// Alignment_25881472926218
// MI455X (gfx1250) — compile-verified, matrix operand order checked by reading, not run on hardware
//
#include <hip/hip_runtime.h>
#include <hip/hip_bf16.h>

// ---------------------------------------------------------------------------
// Model dims
// ---------------------------------------------------------------------------
#define BB   512
#define LL   160
#define RR   160
#define DW   300
#define DA   50
#define DD   350
#define DPAD 352          // 11 * 32, padded K for WMMA f16 K=32 chunks
#define NCH  100
#define CHPAD 128         // 4 x 32 column tiles for 2x2 register blocking
#define FEAT 600

typedef __attribute__((ext_vector_type(16))) _Float16 v16h;
typedef __attribute__((ext_vector_type(8)))  float    v8f;
typedef __attribute__((ext_vector_type(8)))  _Float16 h8v;

// ---------------------------------------------------------------------------
// WMMA 16-bit A/B fragment load from a row-major [16 x K] tile.
// Per CDNA5 ISA layout for 16-bit 16x32 A (and symmetric B):
//   lanes 0-15  : row = lane,    K = 0..7 (VGPR0-3) and 16..23 (VGPR4-7)
//   lanes 16-31 : row = lane-16, K = 8..15           and 24..31
// Each lane therefore loads two contiguous 16-byte chunks.
// ---------------------------------------------------------------------------
__device__ __forceinline__ v16h load_frag16(const _Float16* __restrict__ base, int ldh) {
    int lane = threadIdx.x & 31;
    int hlf  = lane >> 4;
    int row  = lane & 15;
    const _Float16* p = base + (size_t)row * ldh + hlf * 8;
    h8v lo = *(const h8v*)(p);
    h8v hi = *(const h8v*)(p + 16);
    v16h f;
#pragma unroll
    for (int i = 0; i < 8; ++i) { f[i] = lo[i]; f[i + 8] = hi[i]; }
    return f;
}

__device__ __forceinline__ v8f wmma_f16(v16h a, v16h b, v8f c) {
    return __builtin_amdgcn_wmma_f32_16x16x32_f16(
        false, a, false, b, (short)0, c, false, false);
}

// ---------------------------------------------------------------------------
// Embedding gather: l_vec / r_vec as f16, layout [b][pos][DPAD]
// ---------------------------------------------------------------------------
__global__ __launch_bounds__(128) void embed_kernel(
    const int* __restrict__ ltok, const int* __restrict__ lattr,
    const int* __restrict__ rtok, const int* __restrict__ rattr,
    const float* __restrict__ wemb, const float* __restrict__ aemb,
    _Float16* __restrict__ lv, _Float16* __restrict__ rv) {
    int row  = blockIdx.x;                 // 0 .. B*(L+R)-1
    int side = (row >= BB * LL) ? 1 : 0;
    int rr   = side ? (row - BB * LL) : row;
    int b = rr / LL, p = rr % LL;
    const int* tok = side ? rtok : ltok;
    const int* atr = side ? rattr : lattr;
    _Float16* out = (side ? rv : lv) + (size_t)rr * DPAD;
    int t = tok[p * BB + b];
    int a = atr[p * BB + b];
    const float* we = wemb + (size_t)t * DW;
    const float* ae = aemb + (size_t)a * DA;
    for (int d = threadIdx.x; d < DPAD; d += 128) {
        float v = 0.f;
        if (d < DA)       v = ae[d];
        else if (d < DD)  v = we[d - DA];
        out[d] = (_Float16)v;
    }
}

// ---------------------------------------------------------------------------
// Attention GEMM: att[b][l][r] = l_vec . r_vec (masked).
// One wave per 32x32 output tile (2x2 fragment register blocking):
// each K-chunk loads 2 A + 2 B fragments and issues 4 WMMAs.
// Stores BOTH att[b][l][r] and attT[b][r][l] so both softmax directions read
// contiguous rows (the transposed store is an 8-float contiguous run/lane).
// ---------------------------------------------------------------------------
__global__ __launch_bounds__(128) void att_gemm_kernel(
    const _Float16* __restrict__ lv, const _Float16* __restrict__ rv,
    const int* __restrict__ llen, const int* __restrict__ rlen,
    float* __restrict__ att, float* __restrict__ attT) {
    int wid = blockIdx.x * 4 + (threadIdx.x >> 5);   // 0 .. 512*25-1
    int b   = wid / 25;
    int rem = wid % 25;
    int lt  = rem / 5, rt = rem % 5;                 // 32-wide tiles

    const _Float16* A0 = lv + ((size_t)b * LL + lt * 32) * DPAD;
    const _Float16* B0 = rv + ((size_t)b * RR + rt * 32) * DPAD;
    v8f acc00 = {}, acc01 = {}, acc10 = {}, acc11 = {};
#pragma unroll
    for (int kc = 0; kc < DPAD / 32; ++kc) {
        v16h a0 = load_frag16(A0 + kc * 32, DPAD);
        v16h a1 = load_frag16(A0 + (size_t)16 * DPAD + kc * 32, DPAD);
        v16h b0 = load_frag16(B0 + kc * 32, DPAD);
        v16h b1 = load_frag16(B0 + (size_t)16 * DPAD + kc * 32, DPAD);
        acc00 = wmma_f16(a0, b0, acc00);
        acc01 = wmma_f16(a0, b1, acc01);
        acc10 = wmma_f16(a1, b0, acc10);
        acc11 = wmma_f16(a1, b1, acc11);
    }
    int lane = threadIdx.x & 31;
    int hlf  = lane >> 4, col = lane & 15;
    int Llen = llen[b], Rlen = rlen[b];
    const v8f accs[4] = {acc00, acc01, acc10, acc11};
#pragma unroll
    for (int mi = 0; mi < 2; ++mi) {
#pragma unroll
        for (int ni = 0; ni < 2; ++ni) {
            v8f acc = accs[mi * 2 + ni];
            int r   = rt * 32 + ni * 16 + col;
            int l0  = lt * 32 + mi * 16;
            float rmv = (r < Rlen) ? 1.f : 0.f;
            float* out  = att  + ((size_t)b * LL + l0) * RR + r;
            float* outT = attT + ((size_t)b * RR + r) * LL + l0;
#pragma unroll
            for (int i = 0; i < 8; ++i) {
                int m = i + 8 * hlf;
                float v = acc[i] * rmv * (((l0 + m) < Llen) ? 1.f : 0.f);
                out[(size_t)m * RR] = v;   // att[b][l][r]
                outT[m]             = v;   // attT[b][r][l] (contiguous per lane)
            }
        }
    }
}

// ---------------------------------------------------------------------------
// attend: per (b, s) row of a row-major score matrix [b][self][other] —
// masked softmax over 'other', attention stats, top-3 via wave argmax,
// fused counterpart gather and c = w * |v_self - counter| * mask (f16 out).
// One wave per row. Used for both directions (att vs attT).
// ---------------------------------------------------------------------------
__global__ __launch_bounds__(128) void attend_kernel(
    const float* __restrict__ attRows, const _Float16* __restrict__ vself,
    const _Float16* __restrict__ vother, const int* __restrict__ len_self_a,
    const int* __restrict__ len_other_a, _Float16* __restrict__ cout) {
    int wid = blockIdx.x * 4 + (threadIdx.x >> 5);
    int b = wid / LL, s = wid % LL;
    int lane = threadIdx.x & 31;
    int Ls = len_self_a[b], Lo = len_other_a[b];
    const float* arow = attRows + ((size_t)b * LL + s) * RR;

    float vals[5]; int idxs[5];
    float m = -1e30f;
#pragma unroll
    for (int it = 0; it < 5; ++it) {
        int o = it * 32 + lane;
        float v = (o < Lo) ? arow[o] : -1e30f;
        vals[it] = v; idxs[it] = o;
        m = fmaxf(m, v);
    }
    for (int off = 16; off; off >>= 1) m = fmaxf(m, __shfl_xor(m, off, 32));
    float sum = 0.f;
#pragma unroll
    for (int it = 0; it < 5; ++it) {
        int o = it * 32 + lane;
        float e = (o < Lo) ? __expf(vals[it] - m) : 0.f;
        vals[it] = e; sum += e;
    }
    for (int off = 16; off; off >>= 1) sum += __shfl_xor(sum, off, 32);
    float inv = 1.f / (sum + 1e-13f);
    float e2 = 0.f;
#pragma unroll
    for (int it = 0; it < 5; ++it) { vals[it] *= inv; e2 += vals[it] * vals[it]; }
    for (int off = 16; off; off >>= 1) e2 += __shfl_xor(e2, off, 32);

    float lenf = (float)Lo;
    float mean = 1.f / lenf;                         // probs sum to 1
    float w = (e2 / lenf - mean * mean) / fmaxf(mean, 0.001f);

    // top-3 by repeated wave argmax
    float ta[3]; int ti[3];
#pragma unroll
    for (int k = 0; k < 3; ++k) {
        float bv = -1.f; int bi = 0;
#pragma unroll
        for (int it = 0; it < 5; ++it)
            if (vals[it] > bv) { bv = vals[it]; bi = idxs[it]; }
        for (int off = 16; off; off >>= 1) {
            float ov = __shfl_xor(bv, off, 32);
            int   oi = __shfl_xor(bi, off, 32);
            if (ov > bv || (ov == bv && oi < bi)) { bv = ov; bi = oi; }
        }
        ta[k] = bv; ti[k] = bi;
#pragma unroll
        for (int it = 0; it < 5; ++it)
            if (idxs[it] == bi) vals[it] = 0.f;
    }
    float ainv = 1.f / fmaxf(ta[0] + ta[1] + ta[2], 0.001f);
    float a0 = ta[0] * ainv, a1 = ta[1] * ainv, a2 = ta[2] * ainv;
    bool svalid = (s < Ls);

    const _Float16* vrow = vself + ((size_t)b * LL + s) * DPAD;
    const _Float16* g0 = vother + ((size_t)b * RR + ti[0]) * DPAD;
    const _Float16* g1 = vother + ((size_t)b * RR + ti[1]) * DPAD;
    const _Float16* g2 = vother + ((size_t)b * RR + ti[2]) * DPAD;
    _Float16* crow = cout + ((size_t)b * LL + s) * DPAD;
#pragma unroll 1
    for (int d = lane; d < DPAD; d += 32) {
        float x = (float)vrow[d];
        float c = a0 * (float)g0[d] + a1 * (float)g1[d] + a2 * (float)g2[d];
        crow[d] = (_Float16)(svalid ? (w * fabsf(x - c)) : 0.f);
    }
}

// ---------------------------------------------------------------------------
// Conv weight repack: (O=100, D=350, k) f32 -> [ch_pad=128][tap][dpad=352] f16
// so B fragments load exactly like A fragments (row-major over K).
// ---------------------------------------------------------------------------
__global__ void wt_prep_kernel(const float* __restrict__ w1,
                               const float* __restrict__ w2,
                               const float* __restrict__ w3,
                               _Float16* __restrict__ wt) {
    int idx = blockIdx.x * blockDim.x + threadIdx.x;
    const int off1 = CHPAD * DPAD;             // kk=1 region
    const int off2 = off1 + CHPAD * DPAD * 2;  // kk=2 region
    const int tot  = off2 + CHPAD * DPAD * 3;  // kk=3 region end
    if (idx >= tot) return;
    int kk, rel; const float* w;
    if (idx < off1)      { kk = 1; rel = idx;        w = w1; }
    else if (idx < off2) { kk = 2; rel = idx - off1; w = w2; }
    else                 { kk = 3; rel = idx - off2; w = w3; }
    int ldw = kk * DPAD;
    int ch  = rel / ldw;
    int k   = rel % ldw;
    int tap = k / DPAD, d = k % DPAD;
    float v = 0.f;
    if (ch < NCH && d < DD)
        v = w[(size_t)ch * DD * kk + (size_t)d * kk + tap];
    wt[idx] = (_Float16)v;
}

// ---------------------------------------------------------------------------
// Conv-as-GEMM with fused bias + ReLU + masked max-pool (atomicMax on bits).
// One wave per 32x32 tile (2x2 register blocking); KK is compile-time so the
// K loop trip count (KK*11) is known and the compiler can pipeline loads.
// ---------------------------------------------------------------------------
template <int KK>
__global__ __launch_bounds__(128) void conv_gemm_kernel(
    const _Float16* __restrict__ xc, const _Float16* __restrict__ wt,
    const float* __restrict__ bias, const int* __restrict__ len_a,
    float* __restrict__ feats, int featOff) {
    int wid = blockIdx.x * 4 + (threadIdx.x >> 5);
    int rt = wid / (CHPAD / 32);               // 32-row tile
    int ct = wid % (CHPAD / 32);               // 32-channel tile
    const int ldw = KK * DPAD;
    const _Float16* B0 = wt + (size_t)(ct * 32) * ldw;
    size_t row0 = (size_t)rt * 32;

    v8f acc00 = {}, acc01 = {}, acc10 = {}, acc11 = {};
#pragma unroll 2
    for (int kc = 0; kc < KK * (DPAD / 32); ++kc) {
        int k   = kc * 32;
        int tap = k / DPAD;
        int d0  = k - tap * DPAD;
        v16h a0 = load_frag16(xc + (row0 + tap) * DPAD + d0, DPAD);
        v16h a1 = load_frag16(xc + (row0 + 16 + tap) * DPAD + d0, DPAD);
        v16h b0 = load_frag16(B0 + k, ldw);
        v16h b1 = load_frag16(B0 + (size_t)16 * ldw + k, ldw);
        acc00 = wmma_f16(a0, b0, acc00);
        acc01 = wmma_f16(a0, b1, acc01);
        acc10 = wmma_f16(a1, b0, acc10);
        acc11 = wmma_f16(a1, b1, acc11);
    }
    int lane = threadIdx.x & 31;
    int hlf  = lane >> 4, col = lane & 15;
    const v8f accs[4] = {acc00, acc01, acc10, acc11};
#pragma unroll
    for (int mi = 0; mi < 2; ++mi) {
#pragma unroll
        for (int ni = 0; ni < 2; ++ni) {
            v8f acc = accs[mi * 2 + ni];
            int ch  = ct * 32 + ni * 16 + col;
            float bv = (ch < NCH) ? bias[ch] : 0.f;
#pragma unroll
            for (int i = 0; i < 8; ++i) {
                int m    = i + 8 * hlf;
                int grow = (int)row0 + mi * 16 + m;
                int b = grow / LL, t = grow % LL;
                if (ch < NCH && (t + KK) <= len_a[b]) {
                    float v = fmaxf(acc[i] + bv, 0.f);   // relu output >= 0
                    atomicMax((int*)&feats[(size_t)b * FEAT + featOff + ch],
                              __float_as_int(v));
                }
            }
        }
    }
}

__global__ void zero_kernel(float* __restrict__ p, int n) {
    int i = blockIdx.x * blockDim.x + threadIdx.x;
    if (i < n) p[i] = 0.f;
}

// ---------------------------------------------------------------------------
// Dense head: 600 -> 60 (relu) -> 2, one block per batch row.
// ---------------------------------------------------------------------------
__global__ __launch_bounds__(64) void dense_kernel(
    const float* __restrict__ feats, const float* __restrict__ dw1,
    const float* __restrict__ db1, const float* __restrict__ dw2,
    const float* __restrict__ db2, float* __restrict__ out) {
    __shared__ float h[60];
    int b = blockIdx.x, j = threadIdx.x;
    const float* f = feats + (size_t)b * FEAT;
    if (j < 60) {
        float s = db1[j];
        for (int k = 0; k < FEAT; ++k) s += f[k] * dw1[k * 60 + j];
        h[j] = fmaxf(s, 0.f);
    }
    __syncthreads();
    if (j < 2) {
        float s = db2[j];
        for (int k = 0; k < 60; ++k) s += h[k] * dw2[k * 2 + j];
        out[b * 2 + j] = s;
    }
}

// ---------------------------------------------------------------------------
extern "C" void kernel_launch(void* const* d_in, const int* in_sizes, int n_in,
                              void* d_out, int out_size, void* d_ws, size_t ws_size,
                              hipStream_t stream) {
    const int*   l_tokens  = (const int*)d_in[0];
    const int*   l_attrs   = (const int*)d_in[1];
    const int*   l_lengths = (const int*)d_in[2];
    const int*   r_tokens  = (const int*)d_in[3];
    const int*   r_attrs   = (const int*)d_in[4];
    const int*   r_lengths = (const int*)d_in[5];
    const float* word_emb  = (const float*)d_in[6];
    const float* attr_emb  = (const float*)d_in[7];
    const float* conv_w1   = (const float*)d_in[8];
    const float* conv_b1   = (const float*)d_in[9];
    const float* conv_w2   = (const float*)d_in[10];
    const float* conv_b2   = (const float*)d_in[11];
    const float* conv_w3   = (const float*)d_in[12];
    const float* conv_b3   = (const float*)d_in[13];
    const float* dense_w1  = (const float*)d_in[14];
    const float* dense_b1  = (const float*)d_in[15];
    const float* dense_w2  = (const float*)d_in[16];
    const float* dense_b2  = (const float*)d_in[17];
    float* out = (float*)d_out;

    // --- workspace layout (256B aligned slabs) ---
    char* ws = (char*)d_ws;
    size_t off = 0;
    auto carve = [&](size_t bytes) -> char* {
        char* p = ws + off;
        off = (off + bytes + 255) & ~(size_t)255;
        return p;
    };
    const size_t ROWS  = (size_t)BB * LL;            // 81920
    const size_t VEC_B = (ROWS + 2) * DPAD * sizeof(_Float16); // +2 pad rows for conv windows
    _Float16* lv    = (_Float16*)carve(VEC_B);
    _Float16* rv    = (_Float16*)carve(VEC_B);
    _Float16* lc    = (_Float16*)carve(VEC_B);
    _Float16* rc    = (_Float16*)carve(VEC_B);
    float*    att   = (float*)carve((size_t)BB * LL * RR * sizeof(float));
    float*    attT  = (float*)carve((size_t)BB * LL * RR * sizeof(float));
    _Float16* wt    = (_Float16*)carve((size_t)CHPAD * DPAD * 6 * sizeof(_Float16));
    float*    feats = (float*)carve((size_t)BB * FEAT * sizeof(float));
    (void)ws_size; (void)in_sizes; (void)n_in; (void)out_size;

    const int wtOff1 = 0;
    const int wtOff2 = CHPAD * DPAD;
    const int wtOff3 = wtOff2 + CHPAD * DPAD * 2;

    // 1) embeddings -> f16 padded vectors
    embed_kernel<<<BB * (LL + RR), 128, 0, stream>>>(
        l_tokens, l_attrs, r_tokens, r_attrs, word_emb, attr_emb, lv, rv);

    // 2) attention GEMM (WMMA): 512 * 5 * 5 tiles of 32x32, 4 waves/block
    //    writes att and its transpose so both softmax passes are coalesced
    att_gemm_kernel<<<(BB * 5 * 5) / 4, 128, 0, stream>>>(
        lv, rv, l_lengths, r_lengths, att, attT);

    // 3) softmax/top-3/counterpart feature rows (one wave per row), both dirs
    attend_kernel<<<(BB * LL) / 4, 128, 0, stream>>>(
        att,  lv, rv, l_lengths, r_lengths, lc);
    attend_kernel<<<(BB * RR) / 4, 128, 0, stream>>>(
        attT, rv, lv, r_lengths, l_lengths, rc);

    // 4) repack conv weights, zero features
    {
        int tot = CHPAD * DPAD * 6;
        wt_prep_kernel<<<(tot + 255) / 256, 256, 0, stream>>>(
            conv_w1, conv_w2, conv_w3, wt);
        int nf = BB * FEAT;
        zero_kernel<<<(nf + 255) / 256, 256, 0, stream>>>(feats, nf);
    }

    // 5) six conv GEMMs (WMMA) with fused relu + masked max-pool
    {
        const int rowTiles = (BB * LL) / 32;          // 2560
        const int chTiles  = CHPAD / 32;              // 4
        const int blocks   = (rowTiles * chTiles) / 4;
        conv_gemm_kernel<1><<<blocks, 128, 0, stream>>>(lc, wt + wtOff1, conv_b1, l_lengths, feats, 0);
        conv_gemm_kernel<2><<<blocks, 128, 0, stream>>>(lc, wt + wtOff2, conv_b2, l_lengths, feats, 100);
        conv_gemm_kernel<3><<<blocks, 128, 0, stream>>>(lc, wt + wtOff3, conv_b3, l_lengths, feats, 200);
        conv_gemm_kernel<1><<<blocks, 128, 0, stream>>>(rc, wt + wtOff1, conv_b1, r_lengths, feats, 300);
        conv_gemm_kernel<2><<<blocks, 128, 0, stream>>>(rc, wt + wtOff2, conv_b2, r_lengths, feats, 400);
        conv_gemm_kernel<3><<<blocks, 128, 0, stream>>>(rc, wt + wtOff3, conv_b3, r_lengths, feats, 500);
    }

    // 6) dense head
    dense_kernel<<<BB, 64, 0, stream>>>(feats, dense_w1, dense_b1, dense_w2, dense_b2, out);
}
